// LocalAttentionHead_21354577396443
// MI455X (gfx1250) — matrix operand order repaired and checked by reading, not compiled, not run on hardware
//
#include <hip/hip_runtime.h>
#include <hip/hip_bf16.h>

#define D_MODEL 1024
#define D_HEAD  64
#define WINDOW  256
#define SEQ_T   4096
#define BATCH   4
#define NTOK    (BATCH * SEQ_T)
#define WN      (D_HEAD * D_MODEL)   // elements per weight matrix (65536)

typedef __attribute__((ext_vector_type(16))) __bf16 v16bf;
typedef __attribute__((ext_vector_type(8)))  __bf16 v8bf;
typedef __attribute__((ext_vector_type(8)))  float  v8f;

static __device__ __forceinline__ v8f wmma_bf16(v16bf a, v16bf b, v8f c) {
  // D = A(16x32 bf16) * B(32x16 bf16) + C(16x16 f32)
  return __builtin_amdgcn_wmma_f32_16x16x32_bf16(
      /*neg_a=*/false, a, /*neg_b=*/false, b,
      /*c_mod=*/(short)0, c, /*reuse_a=*/false, /*reuse_b=*/false);
}

// A-layout fragment: elements 0..7 from p0[0..7], 8..15 from p1[0..7] (bf16 source)
static __device__ __forceinline__ v16bf load_a_bf16(const __bf16* p0, const __bf16* p1) {
  v8bf lo = *(const v8bf*)p0;
  v8bf hi = *(const v8bf*)p1;
  v16bf r;
#pragma unroll
  for (int j = 0; j < 8; ++j) { r[j] = lo[j]; r[8 + j] = hi[j]; }
  return r;
}

// B-layout fragment: 16 contiguous bf16 (two b128 loads)
static __device__ __forceinline__ v16bf load_b_bf16(const __bf16* p) {
  return load_a_bf16(p, p + 8);
}

// A-layout fragment from fp32 memory (convert to bf16 in regs)
static __device__ __forceinline__ v16bf load_a_f32(const float* p0, const float* p1) {
  float4 a0 = *(const float4*)(p0);
  float4 a1 = *(const float4*)(p0 + 4);
  float4 b0 = *(const float4*)(p1);
  float4 b1 = *(const float4*)(p1 + 4);
  v16bf r;
  r[0]  = (__bf16)a0.x; r[1]  = (__bf16)a0.y; r[2]  = (__bf16)a0.z; r[3]  = (__bf16)a0.w;
  r[4]  = (__bf16)a1.x; r[5]  = (__bf16)a1.y; r[6]  = (__bf16)a1.z; r[7]  = (__bf16)a1.w;
  r[8]  = (__bf16)b0.x; r[9]  = (__bf16)b0.y; r[10] = (__bf16)b0.z; r[11] = (__bf16)b0.w;
  r[12] = (__bf16)b1.x; r[13] = (__bf16)b1.y; r[14] = (__bf16)b1.z; r[15] = (__bf16)b1.w;
  return r;
}

// ---------------------------------------------------------------------------
// Kernel 0: one-shot fp32 -> bf16 conversion of the three weight matrices.
// wb layout: [Wq | Wk | Wv], each row-major [64][1024] bf16 (384 KB total,
// L2-resident for the lifetime of the dispatch).
// ---------------------------------------------------------------------------
__global__ __launch_bounds__(256) void cvt_weights(
    const float* __restrict__ Wq, const float* __restrict__ Wk,
    const float* __restrict__ Wv, __bf16* __restrict__ wb)
{
  const int tid = blockIdx.x * blockDim.x + threadIdx.x;  // 0 .. 3*WN/4-1
  const int m   = tid / (WN / 4);                         // matrix index
  const int e   = (tid - m * (WN / 4)) * 4;               // element offset
  const float* src = (m == 0) ? Wq : (m == 1) ? Wk : Wv;
  float4 f = *(const float4*)(src + e);
  __bf16* dst = wb + (size_t)m * WN + e;
  dst[0] = (__bf16)f.x; dst[1] = (__bf16)f.y;
  dst[2] = (__bf16)f.z; dst[3] = (__bf16)f.w;
}

// ---------------------------------------------------------------------------
// Kernel 1: q/k/v = x @ W^T.  One wave per 16-token tile; B fragments come
// pre-converted from bf16 workspace (no per-step cvt, half the load bytes).
//   q (scaled by 1/sqrt(64)) and k stored bf16 row-major [NTOK][64];
//   v stored transposed bf16 [BATCH][64][SEQ_T] for contiguous B-tile loads.
// ---------------------------------------------------------------------------
__global__ __launch_bounds__(256) void qkv_proj_wmma(
    const float* __restrict__ x,
    const __bf16* __restrict__ wb,
    __bf16* __restrict__ qb,
    __bf16* __restrict__ kb,
    __bf16* __restrict__ vtb)
{
  const int lane = threadIdx.x & 31;
  const int wv   = threadIdx.x >> 5;
  const int h    = lane >> 4;       // lane half (A/B layout split)
  const int l    = lane & 15;
  const int tile = blockIdx.x * 8 + wv;
  const int m0   = tile * 16;       // first token of this tile

  const float*  xrow = x + (size_t)(m0 + l) * D_MODEL;
  const __bf16* wq   = wb;
  const __bf16* wk   = wb + WN;
  const __bf16* wvp  = wb + 2 * WN;

  const v8f zero = {0.f, 0.f, 0.f, 0.f, 0.f, 0.f, 0.f, 0.f};
  v8f cq[4], ck[4], cv[4];
#pragma unroll
  for (int nt = 0; nt < 4; ++nt) { cq[nt] = zero; ck[nt] = zero; cv[nt] = zero; }

  for (int k0 = 0; k0 < D_MODEL; k0 += 32) {
    v16bf a = load_a_f32(xrow + k0 + h * 16, xrow + k0 + h * 16 + 8);
    const int bo = k0 + h * 16;
#pragma unroll
    for (int nt = 0; nt < 4; ++nt) {
      const size_t wrow = (size_t)(nt * 16 + l) * D_MODEL + bo;
      cq[nt] = wmma_bf16(a, load_b_bf16(wq  + wrow), cq[nt]);
      ck[nt] = wmma_bf16(a, load_b_bf16(wk  + wrow), ck[nt]);
      cv[nt] = wmma_bf16(a, load_b_bf16(wvp + wrow), cv[nt]);
    }
  }

  const int bi    = m0 / SEQ_T;
  const int tloc0 = m0 - bi * SEQ_T;
#pragma unroll
  for (int r = 0; r < 8; ++r) {
    const int t  = m0 + r + 8 * h;     // global token (C-layout row)
    const int tl = tloc0 + r + 8 * h;  // token within batch
#pragma unroll
    for (int nt = 0; nt < 4; ++nt) {
      const int hc = nt * 16 + l;      // head column (C-layout col)
      qb[(size_t)t * D_HEAD + hc] = (__bf16)(cq[nt][r] * 0.125f);  // fold 1/sqrt(64)
      kb[(size_t)t * D_HEAD + hc] = (__bf16)(ck[nt][r]);
      vtb[((size_t)bi * D_HEAD + hc) * SEQ_T + tl] = (__bf16)(cv[nt][r]);
    }
  }
}

// ---------------------------------------------------------------------------
// Kernel 2: flash-style local attention. One wave per 16-query tile; iterate
// 32-key blocks inside the causal window; per-row online softmax.
// ---------------------------------------------------------------------------
__global__ __launch_bounds__(256) void local_attn_wmma(
    const __bf16* __restrict__ qb,
    const __bf16* __restrict__ kb,
    const __bf16* __restrict__ vtb,
    float* __restrict__ out)
{
  __shared__ alignas(16) __bf16 plds_all[8][16 * 32];
  const int lane = threadIdx.x & 31;
  const int wv   = threadIdx.x >> 5;
  const int h    = lane >> 4;
  const int l    = lane & 15;
  __bf16* plds = plds_all[wv];

  const int tile = blockIdx.x * 8 + wv;
  const int i0   = tile * 16;          // global query token base
  const int bi   = i0 / SEQ_T;
  const int li0  = i0 - bi * SEQ_T;    // local (in-batch) query base
  const size_t kvbase = (size_t)bi * SEQ_T;

  // Q as A-matrix, two K-chunks over d_head=64 (q was pre-scaled by 1/8)
  const __bf16* qrow = qb + (size_t)(i0 + l) * D_HEAD;
  const v16bf aq0 = load_a_bf16(qrow + h * 16,      qrow + h * 16 + 8);
  const v16bf aq1 = load_a_bf16(qrow + 32 + h * 16, qrow + 40 + h * 16);

  const v8f zero = {0.f, 0.f, 0.f, 0.f, 0.f, 0.f, 0.f, 0.f};
  v8f o[4];
#pragma unroll
  for (int nt = 0; nt < 4; ++nt) o[nt] = zero;
  float mrow[8], lrow[8];
#pragma unroll
  for (int r = 0; r < 8; ++r) { mrow[r] = -1.0e30f; lrow[r] = 0.f; }

  int jstart = li0 - WINDOW;
  if (jstart < 0) jstart = 0;
  jstart &= ~31;
  const int jend = li0 + 15;  // inclusive last key needed by this tile

  for (int j0 = jstart; j0 <= jend; j0 += 32) {
    // ---- scores: S = Q * K^T   (two 16x16 N-tiles of keys) ----
    v8f s[2];
#pragma unroll
    for (int nt = 0; nt < 2; ++nt) {
      const __bf16* krow = kb + (kvbase + (size_t)(j0 + nt * 16 + l)) * D_HEAD;
      v16bf b0 = load_b_bf16(krow + h * 16);       // d_head 0..31
      v16bf b1 = load_b_bf16(krow + 32 + h * 16);  // d_head 32..63
      s[nt] = wmma_bf16(aq0, b0, zero);
      s[nt] = wmma_bf16(aq1, b1, s[nt]);
    }

    // ---- causal+window mask, online softmax over this 32-key block ----
    const int col0 = j0 + l;
    const int col1 = j0 + 16 + l;
    float corr[8];
#pragma unroll
    for (int r = 0; r < 8; ++r) {
      const int row = li0 + r + 8 * h;
      const bool v0 = (col0 <= row) && (col0 >= row - WINDOW);
      const bool v1 = (col1 <= row) && (col1 >= row - WINDOW);
      const float s0 = v0 ? s[0][r] : -1.0e30f;
      const float s1 = v1 ? s[1][r] : -1.0e30f;
      float rm = fmaxf(s0, s1);
      rm = fmaxf(rm, __shfl_xor(rm, 1));
      rm = fmaxf(rm, __shfl_xor(rm, 2));
      rm = fmaxf(rm, __shfl_xor(rm, 4));
      rm = fmaxf(rm, __shfl_xor(rm, 8));
      const float mnew = fmaxf(mrow[r], rm);
      const float c    = __expf(mrow[r] - mnew);
      const float p0   = v0 ? __expf(s0 - mnew) : 0.f;
      const float p1   = v1 ? __expf(s1 - mnew) : 0.f;
      float ps = p0 + p1;
      ps += __shfl_xor(ps, 1);
      ps += __shfl_xor(ps, 2);
      ps += __shfl_xor(ps, 4);
      ps += __shfl_xor(ps, 8);
      lrow[r] = lrow[r] * c + ps;
      mrow[r] = mnew;
      corr[r] = c;
      // C-layout -> row-major P in LDS (bf16), for A-layout reload
      plds[(r + 8 * h) * 32 + l]      = (__bf16)p0;
      plds[(r + 8 * h) * 32 + 16 + l] = (__bf16)p1;
    }

    // rescale running output by exp(m_old - m_new)
#pragma unroll
    for (int nt = 0; nt < 4; ++nt)
#pragma unroll
      for (int r = 0; r < 8; ++r) o[nt][r] *= corr[r];

    asm volatile("s_wait_dscnt 0" ::: "memory");  // LDS RAW (within-wave transpose)

    // ---- O += P * V  (P as A 16x32; V via transposed layout, 4 head tiles) ----
    const v16bf ap = load_a_bf16(plds + l * 32 + h * 16, plds + l * 32 + h * 16 + 8);
#pragma unroll
    for (int nt = 0; nt < 4; ++nt) {
      const __bf16* vrow = vtb + ((size_t)bi * D_HEAD + nt * 16 + l) * SEQ_T + j0;
      o[nt] = wmma_bf16(ap, load_b_bf16(vrow + h * 16), o[nt]);
    }
    asm volatile("" ::: "memory");  // keep next block's LDS stores behind these loads
  }

  // ---- normalize and store ----
#pragma unroll
  for (int r = 0; r < 8; ++r) {
    const int t = i0 + r + 8 * h;
    const float inv = 1.0f / lrow[r];
#pragma unroll
    for (int nt = 0; nt < 4; ++nt)
      out[(size_t)t * D_HEAD + nt * 16 + l] = o[nt][r] * inv;
  }
}

extern "C" void kernel_launch(void* const* d_in, const int* in_sizes, int n_in,
                              void* d_out, int out_size, void* d_ws, size_t ws_size,
                              hipStream_t stream) {
  const float* x  = (const float*)d_in[0];
  const float* Wq = (const float*)d_in[1];
  const float* Wk = (const float*)d_in[2];
  const float* Wv = (const float*)d_in[3];
  float* out = (float*)d_out;

  // workspace: wb (384KB bf16 weights) | q (2MB) | k (2MB) | v^T (2MB)
  __bf16* wb  = reinterpret_cast<__bf16*>(d_ws);
  __bf16* qb  = wb + (size_t)3 * WN;
  __bf16* kb  = qb + (size_t)NTOK * D_HEAD;
  __bf16* vtb = kb + (size_t)NTOK * D_HEAD;

  const int ntiles = NTOK / 16;   // 1024 16-token tiles
  const int blocks = ntiles / 8;  // 8 waves (256 threads) per block

  cvt_weights<<<(3 * WN / 4) / 256, 256, 0, stream>>>(Wq, Wk, Wv, wb);
  qkv_proj_wmma<<<blocks, 256, 0, stream>>>(x, wb, qb, kb, vtb);
  local_attn_wmma<<<blocks, 256, 0, stream>>>(qb, kb, vtb, out);

  (void)in_sizes; (void)n_in; (void)out_size; (void)ws_size;
}
